// CrossAttention_23639499997334
// MI455X (gfx1250) — compile-verified, matrix operand order checked by reading, not run on hardware
//
#include <hip/hip_runtime.h>
#include <hip/hip_bf16.h>
#include <stdint.h>

// Problem constants (reference): B=2, S=2048, D=1024, H=16, Dh=64
#define BATCH   2
#define SEQ     2048
#define D_EMBED 1024
#define NHEADS  16
#define HDIM    64

typedef __bf16 v16bf __attribute__((ext_vector_type(16)));
typedef __bf16 v8bf  __attribute__((ext_vector_type(8)));
typedef float  v8f   __attribute__((ext_vector_type(8)));
typedef unsigned int u32x4 __attribute__((ext_vector_type(4)));
typedef int          i32x8 __attribute__((ext_vector_type(8)));
typedef int          i32x4 __attribute__((ext_vector_type(4)));

union FragU { v16bf v; v8bf h[2]; };

// Load a 16-lane-row x 32-K WMMA operand fragment from a row-major buffer.
// p points at (row base + k-window start); hi8 = 8*(lane/16).
// Per CDNA5 ISA 7.12.2: lane holds elements [hi8..hi8+7] and [hi8+16..hi8+23].
__device__ __forceinline__ v16bf ld_frag(const __bf16* p, unsigned hi8) {
  FragU f;
  f.h[0] = *(const v8bf*)(p + hi8);
  f.h[1] = *(const v8bf*)(p + hi8 + 16);
  return f.v;
}

__device__ __forceinline__ v8f wmma_bf16(v16bf a, v16bf b, v8f c) {
  return __builtin_amdgcn_wmma_f32_16x16x32_bf16(false, a, false, b,
                                                 (short)0, c, false, false);
}

#if defined(__has_builtin)
#if __has_builtin(__builtin_amdgcn_tensor_load_to_lds) && __has_builtin(__builtin_amdgcn_s_wait_tensorcnt)
#define HAVE_TDM 1
#endif
#endif
#ifndef HAVE_TDM
#define HAVE_TDM 0
#endif

#if HAVE_TDM
#define TENSOR_WAIT(n) __builtin_amdgcn_s_wait_tensorcnt(n)
#else
#define TENSOR_WAIT(n) ((void)0)
#endif

// Issue a 2D TDM tile load (bf16 elements) into LDS. Executed by one wave.
// D# packing per CDNA5 ISA 8.3/8.4 (group0: count/lds_addr/global_addr/type=2;
// group1: data_size=2B, tensor dims, tile dims, dim0 stride).
// amdgpu-toolchain (clang-23) builtin form: 6 args
//   (u32x4 g0, i32x8 g1, i32x4 g2, i32x4 g3, i32x8 pad, i32 cpol)
__device__ __forceinline__ void tdm_load_2d(void* lds_ptr, const void* gptr,
                                            unsigned tile_d0, unsigned tile_d1,
                                            unsigned tensor_d0, unsigned tensor_d1,
                                            unsigned long long row_stride_elems) {
#if HAVE_TDM
  unsigned long long ga = (unsigned long long)(uintptr_t)gptr;
  unsigned lds_off = (unsigned)(uintptr_t)lds_ptr;  // low 32 bits of flat addr = LDS offset
  u32x4 g0;
  g0.x = 1u;                                   // count=1, user descriptor
  g0.y = lds_off;                              // lds_addr
  g0.z = (unsigned)(ga & 0xFFFFFFFFull);       // global_addr[31:0]
  g0.w = (unsigned)((ga >> 32) & 0x01FFFFFFull) | (2u << 30);  // addr[56:32] | type=2
  i32x8 g1;
  g1[0] = (int)(1u << 16);                                     // data_size=1 (2 bytes)
  g1[1] = (int)((tensor_d0 & 0xFFFFu) << 16);                  // tensor_dim0[15:0]
  g1[2] = (int)((tensor_d0 >> 16) | ((tensor_d1 & 0xFFFFu) << 16));
  g1[3] = (int)((tensor_d1 >> 16) | (tile_d0 << 16));          // tile_dim0
  g1[4] = (int)(tile_d1 & 0xFFFFu);                            // tile_dim1 (tile_dim2=0)
  g1[5] = (int)(row_stride_elems & 0xFFFFFFFFull);             // tensor_dim0_stride lo
  g1[6] = (int)((row_stride_elems >> 32) & 0xFFFFull);         // stride hi, dim1_stride=0
  g1[7] = 0;
  i32x4 gz4 = (i32x4){0, 0, 0, 0};
  i32x8 gz8 = (i32x8){0, 0, 0, 0, 0, 0, 0, 0};
  __builtin_amdgcn_tensor_load_to_lds(g0, g1, gz4, gz4, gz8, 0);
#else
  // Fallback: cooperative copy by the issuing wave.
  const __bf16* g = (const __bf16*)gptr;
  __bf16* l = (__bf16*)lds_ptr;
  int lane = (int)(threadIdx.x & 31);
  for (unsigned y = 0; y < tile_d1; ++y)
    for (unsigned x = (unsigned)lane; x < tile_d0; x += 32)
      l[y * tile_d0 + x] = g[y * row_stride_elems + x];
#endif
}

// ---------------------------------------------------------------------------
// fp32 -> bf16 elementwise convert
__global__ void cvt_bf16_kernel(const float* __restrict__ x, __bf16* __restrict__ y, int n) {
  for (int i = blockIdx.x * blockDim.x + threadIdx.x; i < n; i += gridDim.x * blockDim.x)
    y[i] = (__bf16)x[i];
}

// ---------------------------------------------------------------------------
// LayerNorm one row of D=1024 per block (256 threads, 8 waves), bf16 output
__global__ __launch_bounds__(256) void ln_bf16_kernel(const float* __restrict__ x,
                                                      const float* __restrict__ w,
                                                      const float* __restrict__ bias,
                                                      __bf16* __restrict__ y) {
  __shared__ float red[2][8];
  const int row = blockIdx.x;
  const float4* xr = (const float4*)(x + (size_t)row * D_EMBED);
  float4 v = xr[threadIdx.x];
  float s = v.x + v.y + v.z + v.w;
  float q = v.x * v.x + v.y * v.y + v.z * v.z + v.w * v.w;
  for (int off = 16; off; off >>= 1) {
    s += __shfl_down(s, off, 32);
    q += __shfl_down(q, off, 32);
  }
  const int wid = threadIdx.x >> 5, lane = threadIdx.x & 31;
  if (lane == 0) { red[0][wid] = s; red[1][wid] = q; }
  __syncthreads();
  if (threadIdx.x == 0) {
    float ts = 0.f, tq = 0.f;
    for (int i = 0; i < 8; ++i) { ts += red[0][i]; tq += red[1][i]; }
    red[0][0] = ts; red[1][0] = tq;
  }
  __syncthreads();
  const float mu = red[0][0] * (1.f / (float)D_EMBED);
  const float var = red[1][0] * (1.f / (float)D_EMBED) - mu * mu;
  const float rstd = rsqrtf(var + 1e-5f);
  const int c = threadIdx.x * 4;
  __bf16* yr = y + (size_t)row * D_EMBED + c;
  const float* wp = w + c;
  const float* bp = bias + c;
  yr[0] = (__bf16)((v.x - mu) * rstd * wp[0] + bp[0]);
  yr[1] = (__bf16)((v.y - mu) * rstd * wp[1] + bp[1]);
  yr[2] = (__bf16)((v.z - mu) * rstd * wp[2] + bp[2]);
  yr[3] = (__bf16)((v.w - mu) * rstd * wp[3] + bp[3]);
}

// ---------------------------------------------------------------------------
// GEMM: dst_head[b][h][s][dh] = scale * sum_k X[m][k] * W[n][k]
// M = B*S tokens (blockIdx.x, 128/block), N = features (blockIdx.y, 128/block)
// 8 waves: 4 M-slices x 2 N-slices; wave tile 32x64 = 2x4 WMMA accumulators.
// A/B tiles (128x32 bf16) TDM-loaded into double-buffered LDS.
__global__ __launch_bounds__(256) void gemm_qkv_kernel(const __bf16* __restrict__ X,
                                                       const __bf16* __restrict__ W,
                                                       __bf16* __restrict__ dst,
                                                       float scale) {
  __shared__ alignas(16) __bf16 At[2][128][32];
  __shared__ alignas(16) __bf16 Bt[2][128][32];
  const int tid = threadIdx.x;
  const int wid = tid >> 5, lane = tid & 31, l16 = lane & 15;
  const unsigned hi8 = (unsigned)((lane >> 4) << 3);
  const int waveM = wid & 3, waveN = wid >> 2;
  const int m0 = blockIdx.x * 128, n0 = blockIdx.y * 128;

  const v8f vz = {0.f, 0.f, 0.f, 0.f, 0.f, 0.f, 0.f, 0.f};
  v8f acc[2][4];
  for (int i = 0; i < 2; ++i)
    for (int j = 0; j < 4; ++j) acc[i][j] = vz;

  const size_t arow = (size_t)m0 * D_EMBED;
  const size_t brow = (size_t)n0 * D_EMBED;
  if (wid == 0) {
    tdm_load_2d(&At[0][0][0], X + arow, 32, 128, D_EMBED, 1u << 20, D_EMBED);
    tdm_load_2d(&Bt[0][0][0], W + brow, 32, 128, D_EMBED, 1u << 20, D_EMBED);
  }
  const int KSTEPS = D_EMBED / 32;
  for (int ks = 0; ks < KSTEPS; ++ks) {
    const int cur = ks & 1;
    if (wid == 0) {
      if (ks + 1 < KSTEPS) {
        const int k1 = (ks + 1) * 32;
        tdm_load_2d(&At[cur ^ 1][0][0], X + arow + k1, 32, 128, D_EMBED, 1u << 20, D_EMBED);
        tdm_load_2d(&Bt[cur ^ 1][0][0], W + brow + k1, 32, 128, D_EMBED, 1u << 20, D_EMBED);
        TENSOR_WAIT(2);  // current buffer's two loads complete (in-order)
      } else {
        TENSOR_WAIT(0);
      }
    }
    __syncthreads();
    v16bf a0 = ld_frag(&At[cur][waveM * 32 + l16][0], hi8);
    v16bf a1 = ld_frag(&At[cur][waveM * 32 + 16 + l16][0], hi8);
#pragma unroll
    for (int nt = 0; nt < 4; ++nt) {
      v16bf bfrag = ld_frag(&Bt[cur][waveN * 64 + nt * 16 + l16][0], hi8);
      acc[0][nt] = wmma_bf16(a0, bfrag, acc[0][nt]);
      acc[1][nt] = wmma_bf16(a1, bfrag, acc[1][nt]);
    }
    __syncthreads();
  }
  // Epilogue: scatter into head-major bf16 [B][H][S][HDIM]
#pragma unroll
  for (int mt = 0; mt < 2; ++mt)
#pragma unroll
    for (int nt = 0; nt < 4; ++nt)
#pragma unroll
      for (int vv = 0; vv < 8; ++vv) {
        const int m = m0 + waveM * 32 + mt * 16 + vv + (int)hi8;
        const int n = n0 + waveN * 64 + nt * 16 + l16;
        const int b = m >> 11, s = m & (SEQ - 1);
        const int h = n >> 6, dh = n & (HDIM - 1);
        const size_t idx = ((((size_t)b * NHEADS + h) * SEQ) + s) * HDIM + dh;
        dst[idx] = (__bf16)(acc[mt][nt][vv] * scale);
      }
}

// ---------------------------------------------------------------------------
// Flash attention: one (b, h, 64-row Q tile) per block, 4 waves x 16 Q rows.
// K tile TDM-loaded; V tile transposed into LDS by all threads; online softmax.
__global__ __launch_bounds__(128) void attn_kernel(const __bf16* __restrict__ Q,
                                                   const __bf16* __restrict__ K,
                                                   const __bf16* __restrict__ V,
                                                   const float* __restrict__ resid,
                                                   float* __restrict__ out) {
  __shared__ alignas(16) __bf16 Kt[64][64];   // [key][dh]
  __shared__ alignas(16) __bf16 Vt[64][80];   // [dh][key], padded rows (160B, 16B-aligned)
  __shared__ alignas(16) __bf16 Pb[4][16][64]; // per-wave P staging [qrow][key]
  const int tid = threadIdx.x;
  const int wid = tid >> 5, lane = tid & 31, l16 = lane & 15;
  const unsigned hi8 = (unsigned)((lane >> 4) << 3);
  const int qt = blockIdx.x, h = blockIdx.y, b = blockIdx.z;
  const size_t bh = ((size_t)b * NHEADS + h) * SEQ;

  // Q fragments held in registers for the whole pass (scale folded in at GEMM)
  const __bf16* qrow = Q + (bh + (size_t)qt * 64 + (size_t)wid * 16 + l16) * HDIM;
  const v16bf aq0 = ld_frag(qrow, hi8);
  const v16bf aq1 = ld_frag(qrow + 32, hi8);

  const v8f vz = {0.f, 0.f, 0.f, 0.f, 0.f, 0.f, 0.f, 0.f};
  v8f o[4];
  for (int i = 0; i < 4; ++i) o[i] = vz;
  float mrow[8], lrow[8];
  for (int i = 0; i < 8; ++i) { mrow[i] = -1e30f; lrow[i] = 0.f; }

  for (int kt = 0; kt < SEQ / 64; ++kt) {
    __syncthreads();  // previous tile fully consumed before overwrite
    const __bf16* kp = K + (bh + (size_t)kt * 64) * HDIM;
    const __bf16* vp = V + (bh + (size_t)kt * 64) * HDIM;
    if (kt + 1 < SEQ / 64) __builtin_prefetch(kp + 64 * HDIM, 0, 1);
    if (wid == 0)
      tdm_load_2d(&Kt[0][0], kp, 64 * HDIM, 1, 64 * HDIM, 1, 64 * HDIM);
    {  // V transpose staging: 128 threads x 32 elems
      const int key = tid >> 1;
      const int dh0 = (tid & 1) * 32;
      const __bf16* vr = vp + key * HDIM + dh0;
#pragma unroll
      for (int j = 0; j < 32; ++j) Vt[dh0 + j][key] = vr[j];
    }
    if (wid == 0) TENSOR_WAIT(0);
    __syncthreads();

    // Scores: S = Q (16x64) x K^T (64x64) -> 4 accumulators of 16x16
    v8f sc[4];
    for (int i = 0; i < 4; ++i) sc[i] = vz;
#pragma unroll
    for (int nt = 0; nt < 4; ++nt) {
      v16bf bk0 = ld_frag(&Kt[nt * 16 + l16][0], hi8);
      sc[nt] = wmma_bf16(aq0, bk0, sc[nt]);
      v16bf bk1 = ld_frag(&Kt[nt * 16 + l16][32], hi8);
      sc[nt] = wmma_bf16(aq1, bk1, sc[nt]);
    }

    // Online softmax per Q row (rows live across 16-lane half-wave groups)
#pragma unroll
    for (int vv = 0; vv < 8; ++vv) {
      float t = fmaxf(fmaxf(sc[0][vv], sc[1][vv]), fmaxf(sc[2][vv], sc[3][vv]));
      t = fmaxf(t, __shfl_xor(t, 1, 32));
      t = fmaxf(t, __shfl_xor(t, 2, 32));
      t = fmaxf(t, __shfl_xor(t, 4, 32));
      t = fmaxf(t, __shfl_xor(t, 8, 32));
      const float nm = fmaxf(mrow[vv], t);
      const float alpha = __expf(mrow[vv] - nm);
      mrow[vv] = nm;
      float ssum = 0.f;
      const int r = vv + (int)hi8;
#pragma unroll
      for (int nt = 0; nt < 4; ++nt) {
        const float p = __expf(sc[nt][vv] - nm);
        ssum += p;
        Pb[wid][r][nt * 16 + l16] = (__bf16)p;
      }
      ssum += __shfl_xor(ssum, 1, 32);
      ssum += __shfl_xor(ssum, 2, 32);
      ssum += __shfl_xor(ssum, 4, 32);
      ssum += __shfl_xor(ssum, 8, 32);
      lrow[vv] = lrow[vv] * alpha + ssum;
      o[0][vv] *= alpha; o[1][vv] *= alpha; o[2][vv] *= alpha; o[3][vv] *= alpha;
    }

    // O += P (16x64) x V (64x64); per-wave Pb region, in-order LDS within wave
    v16bf ap0 = ld_frag(&Pb[wid][l16][0], hi8);
    v16bf ap1 = ld_frag(&Pb[wid][l16][32], hi8);
#pragma unroll
    for (int nt = 0; nt < 4; ++nt) {
      v16bf bv0 = ld_frag(&Vt[nt * 16 + l16][0], hi8);
      o[nt] = wmma_bf16(ap0, bv0, o[nt]);
      v16bf bv1 = ld_frag(&Vt[nt * 16 + l16][32], hi8);
      o[nt] = wmma_bf16(ap1, bv1, o[nt]);
    }
  }

  // Epilogue: normalize, add residual, store fp32 [B][S][D]
#pragma unroll
  for (int nt = 0; nt < 4; ++nt)
#pragma unroll
    for (int vv = 0; vv < 8; ++vv) {
      const int s_idx = qt * 64 + wid * 16 + vv + (int)hi8;
      const int d = h * HDIM + nt * 16 + l16;
      const size_t idx = ((size_t)b * SEQ + s_idx) * D_EMBED + d;
      out[idx] = o[nt][vv] / lrow[vv] + resid[idx];
    }
}

// ---------------------------------------------------------------------------
extern "C" void kernel_launch(void* const* d_in, const int* in_sizes, int n_in,
                              void* d_out, int out_size, void* d_ws, size_t ws_size,
                              hipStream_t stream) {
  const float* inputs = (const float*)d_in[0];   // (B,S,D) fp32, also residual
  const float* cross  = (const float*)d_in[1];   // (B,S,D) fp32
  const float* lnw    = (const float*)d_in[2];   // (D,)
  const float* lnb    = (const float*)d_in[3];   // (D,)
  const float* kvw    = (const float*)d_in[4];   // (2D, D)
  const float* qw     = (const float*)d_in[5];   // (D, D)
  float* out = (float*)d_out;

  char* ws = (char*)d_ws;
  __bf16* x_bf   = (__bf16*)(ws);                         //  8 MB LN(x)
  __bf16* c_bf   = (__bf16*)(ws + (size_t)8  * 1048576);  //  8 MB cross
  __bf16* qw_bf  = (__bf16*)(ws + (size_t)16 * 1048576);  //  2 MB
  __bf16* kvw_bf = (__bf16*)(ws + (size_t)18 * 1048576);  //  4 MB
  __bf16* qh     = (__bf16*)(ws + (size_t)22 * 1048576);  //  8 MB [B][H][S][Dh]
  __bf16* kh     = (__bf16*)(ws + (size_t)30 * 1048576);  //  8 MB
  __bf16* vh     = (__bf16*)(ws + (size_t)38 * 1048576);  //  8 MB  (total 46 MB)

  cvt_bf16_kernel<<<1024, 256, 0, stream>>>(cross, c_bf, BATCH * SEQ * D_EMBED);
  cvt_bf16_kernel<<<256, 256, 0, stream>>>(qw, qw_bf, D_EMBED * D_EMBED);
  cvt_bf16_kernel<<<512, 256, 0, stream>>>(kvw, kvw_bf, 2 * D_EMBED * D_EMBED);
  ln_bf16_kernel<<<BATCH * SEQ, 256, 0, stream>>>(inputs, lnw, lnb, x_bf);

  dim3 ggrid(BATCH * SEQ / 128, D_EMBED / 128);
  // q scaled by 1/sqrt(Dh) here so attention needs no extra scale
  gemm_qkv_kernel<<<ggrid, 256, 0, stream>>>(x_bf, qw_bf, qh, 0.125f);
  gemm_qkv_kernel<<<ggrid, 256, 0, stream>>>(c_bf, kvw_bf, kh, 1.0f);
  gemm_qkv_kernel<<<ggrid, 256, 0, stream>>>(c_bf, kvw_bf + (size_t)D_EMBED * D_EMBED, vh, 1.0f);

  attn_kernel<<<dim3(SEQ / 64, NHEADS, BATCH), 128, 0, stream>>>(qh, kh, vh, inputs, out);
}
